// Net_89730456748751
// MI455X (gfx1250) — matrix operand order repaired and checked by reading, not compiled, not run on hardware
//
#include <hip/hip_runtime.h>
#include <hip/hip_bf16.h>

#define H   128
#define NN  10000
#define NE  10000
#define NG  500
#define NC  40000

typedef _Float16 v16h __attribute__((ext_vector_type(16)));
typedef _Float16 h8   __attribute__((ext_vector_type(8)));
typedef _Float16 h2   __attribute__((ext_vector_type(2)));
typedef float    v8f  __attribute__((ext_vector_type(8)));

__device__ __forceinline__ float sigm(float x){ return 1.f/(1.f+expf(-x)); }

__device__ __forceinline__ void atomicMaxF(float* addr, float val){
  unsigned* ua = (unsigned*)addr;
  unsigned old = __float_as_uint(*addr);
  while (__uint_as_float(old) < val) {
    unsigned assumed = old;
    old = atomicCAS(ua, assumed, __float_as_uint(val));
    if (old == assumed) break;
  }
}

// Fragment-major B layout: for K-block kb (32 rows) and N-tile nt (16 cols),
// the WMMA B fragment (32 lanes x 16 halves, contiguous k = 16*half + s) is stored contiguously:
//   Bf[(((kb*(N/16) + nt)*32 + lane)*16) + 2*v + e]
// where lane = (n%16) + 16*((k%32)/16), v = ((k%32)%16)/2, e = k&1.
__device__ __forceinline__ size_t bfrag_idx(int k, int n, int N){
  const int kb = k >> 5, krem = k & 31;
  const int nt = n >> 4, nr = n & 15;
  const int hl = krem >> 4;
  const int v  = (krem & 15) >> 1;
  const int e  = krem & 1;
  return (((size_t)kb*(N >> 4) + nt)*32 + (nr + 16*hl))*16 + 2*v + e;
}

// ---------------------------------------------------------------------------
// WMMA GEMM: C[MxN] = A[MxK](f16 row-major) @ B (fragment-major, see above).
// One wave computes a 32x32 super-tile (2x2 WMMA tiles): per K-step,
// 4x b128 A loads + 4x b128 B loads feed 4 v_wmma_f32_16x16x32_f16.
// epi: 0 none, 1 BN, 2 BN+ReLU.  Cf(f32) and/or Ch(f16) outputs.
// Requires K%32==0, N%32==0; ragged M handled by row-clamped loads + store
// guards (out-of-range A rows only feed out-of-range C rows, never stored).
// ---------------------------------------------------------------------------
__global__ void gemm_wmma(const _Float16* __restrict__ A,
                          const _Float16* __restrict__ Bf,
                          int M, int K, int N,
                          const float* __restrict__ bg, const float* __restrict__ bb,
                          const float* __restrict__ bm, const float* __restrict__ bv,
                          int epi, float* __restrict__ Cf, _Float16* __restrict__ Ch)
{
  const int wave = threadIdx.x >> 5;
  const int lane = threadIdx.x & 31;
  const int nT = N >> 4;                  // 16-wide N tiles
  const int tN = N >> 5;                  // 32-wide super-tiles
  const int tM = (M + 31) >> 5;
  const long tile = (long)blockIdx.x * 8 + wave;
  if (tile >= (long)tM * tN) return;      // wave-uniform exit: EXEC stays full
  const int tm = (int)(tile / tN);
  const int tn = (int)(tile % tN);
  const int hl = lane >> 4;               // half-lane select
  const int mr = lane & 15;
  const int m0 = tm*32 + mr, m1 = m0 + 16;
  const int n0 = tn*32 + mr, n1 = n0 + 16;
  // row-clamped A addresses: branch-free loads, always in-bounds
  const size_t arow0 = (size_t)(m0 < M ? m0 : M-1) * K;
  const size_t arow1 = (size_t)(m1 < M ? m1 : M-1) * K;
  // per-lane fragment bases in fragment-major B
  const _Float16* bp0 = Bf + ((size_t)(tn*2    )*32 + lane)*16;
  const _Float16* bp1 = Bf + ((size_t)(tn*2 + 1)*32 + lane)*16;
  const size_t bstep = (size_t)nT * 32 * 16;   // halves per K-block

  v8f c00 = {0.f,0.f,0.f,0.f,0.f,0.f,0.f,0.f};
  v8f c01 = c00, c10 = c00, c11 = c00;

  for (int kk = 0; kk < K; kk += 32) {
    // A fragment (16-bit 16x32): lane's halves are two contiguous 16B
    // chunks at k-offsets [16*hl, 16*hl+8) and [16*hl+8, 16*hl+16)
    h8 a0lo = *(const h8*)(A + arow0 + kk +     16*hl);
    h8 a0hi = *(const h8*)(A + arow0 + kk + 8 + 16*hl);
    h8 a1lo = *(const h8*)(A + arow1 + kk +     16*hl);
    h8 a1hi = *(const h8*)(A + arow1 + kk + 8 + 16*hl);
    v16h a0 = __builtin_shufflevector(a0lo, a0hi, 0,1,2,3,4,5,6,7,8,9,10,11,12,13,14,15);
    v16h a1 = __builtin_shufflevector(a1lo, a1hi, 0,1,2,3,4,5,6,7,8,9,10,11,12,13,14,15);
    // B fragments: one contiguous 32B load per lane per N-tile
    const size_t bo = (size_t)(kk >> 5) * bstep;
    v16h b0 = *(const v16h*)(bp0 + bo);
    v16h b1 = *(const v16h*)(bp1 + bo);

    c00 = __builtin_amdgcn_wmma_f32_16x16x32_f16(false, a0, false, b0, (short)0, c00, false, false);
    c01 = __builtin_amdgcn_wmma_f32_16x16x32_f16(false, a0, false, b1, (short)0, c01, false, false);
    c10 = __builtin_amdgcn_wmma_f32_16x16x32_f16(false, a1, false, b0, (short)0, c10, false, false);
    c11 = __builtin_amdgcn_wmma_f32_16x16x32_f16(false, a1, false, b1, (short)0, c11, false, false);
  }

  float g0=1.f, e0=0.f, u0=0.f, s0=1.f, g1v=1.f, e1v=0.f, u1=0.f, s1=1.f;
  if (epi >= 1) {
    g0 = bg[n0]; e0 = bb[n0]; u0 = bm[n0]; s0 = rsqrtf(bv[n0] + 1e-5f);
    g1v= bg[n1]; e1v= bb[n1]; u1 = bm[n1]; s1 = rsqrtf(bv[n1] + 1e-5f);
  }
#pragma unroll
  for (int r = 0; r < 8; ++r) {
    const int row0 = tm*32 + r + 8*hl;    // ISA C/D layout: M = r + 8*half
    const int row1 = row0 + 16;
    float v00 = c00[r], v01 = c01[r], v10 = c10[r], v11 = c11[r];
    if (epi >= 1) {
      v00 = g0 *(v00-u0)*s0 + e0;  v01 = g1v*(v01-u1)*s1 + e1v;
      v10 = g0 *(v10-u0)*s0 + e0;  v11 = g1v*(v11-u1)*s1 + e1v;
      if (epi == 2) { v00=fmaxf(v00,0.f); v01=fmaxf(v01,0.f); v10=fmaxf(v10,0.f); v11=fmaxf(v11,0.f); }
    }
    if (row0 < M) {
      if (Cf) { Cf[(size_t)row0*N + n0] = v00; Cf[(size_t)row0*N + n1] = v01; }
      if (Ch) { Ch[(size_t)row0*N + n0] = (_Float16)v00; Ch[(size_t)row0*N + n1] = (_Float16)v01; }
    }
    if (row1 < M) {
      if (Cf) { Cf[(size_t)row1*N + n0] = v10; Cf[(size_t)row1*N + n1] = v11; }
      if (Ch) { Ch[(size_t)row1*N + n0] = (_Float16)v10; Ch[(size_t)row1*N + n1] = (_Float16)v11; }
    }
  }
}

// ------------------------------- small kernels ------------------------------
__global__ void k_zero(float* p, long n){ long i=(long)blockIdx.x*256+threadIdx.x; if(i<n) p[i]=0.f; }
__global__ void k_fill(float* p, float v, long n){ long i=(long)blockIdx.x*256+threadIdx.x; if(i<n) p[i]=v; }

__global__ void k_cvt_pad_cols(const float* s, _Float16* d, int R, int Cs, int Cd){
  long i=(long)blockIdx.x*256+threadIdx.x; if(i>=(long)R*Cd) return;
  int r=(int)(i/Cd), c=(int)(i%Cd);
  d[i] = (c<Cs) ? (_Float16)s[(size_t)r*Cs+c] : (_Float16)0.f;
}
// pack B (Ks x N row-major f32, zero-padded to K rows) -> fragment-major f16
__global__ void k_pack_b(const float* s, _Float16* d, int K, int N, int Ks){
  long i=(long)blockIdx.x*256+threadIdx.x; if(i>=(long)K*N) return;
  int k=(int)(i/N), n=(int)(i%N);
  float v = (k < Ks) ? s[(size_t)k*N + n] : 0.f;
  d[bfrag_idx(k,n,N)] = (_Float16)v;
}
// pack B = s^T (s is N x K row-major f32) -> fragment-major f16
__global__ void k_pack_bT(const float* s, _Float16* d, int K, int N){
  long i=(long)blockIdx.x*256+threadIdx.x; if(i>=(long)K*N) return;
  int k=(int)(i/N), n=(int)(i%N);
  d[bfrag_idx(k,n,N)] = (_Float16)s[(size_t)n*K + k];
}

// edge encoder 1: K=6 -> VALU
__global__ void k_ee1(const float* __restrict__ E, const float* __restrict__ W,
                      const float* g, const float* b, const float* m, const float* v,
                      _Float16* out){
  long i=(long)blockIdx.x*256+threadIdx.x; if(i>=(long)NE*256) return;
  int row=(int)(i>>8), col=(int)(i&255);
  float s=0.f;
#pragma unroll
  for(int c=0;c<6;++c) s += E[(size_t)row*6+c]*W[c*256+col];
  s = g[col]*(s-m[col])*rsqrtf(v[col]+1e-5f)+b[col];
  out[i] = (_Float16)fmaxf(s,0.f);
}

__global__ void k_deg(const int* ei, float* cnt_src, float* cnt_dst){
  int e=blockIdx.x*256+threadIdx.x; if(e>=NE) return;
  atomicAdd(&cnt_src[ei[2*e]],1.f);
  atomicAdd(&cnt_dst[ei[2*e+1]],1.f);
}

__global__ void k_init_state(const float* x, float* hid, _Float16* hid_h, float* acc){
  long i=(long)blockIdx.x*256+threadIdx.x; if(i>=(long)NN*H) return;
  float v=x[i]; hid[i]=v; hid_h[i]=(_Float16)v; acc[i]=0.f;
}
__global__ void k_acc(const float* hid, const float* x, float* acc){
  long i=(long)blockIdx.x*256+threadIdx.x; if(i<(long)NN*H) acc[i]+=hid[i]+x[i];
}

// per-edge einsum: msg_e = nd[src] @ M[e] (f16 128x128 per edge), scatter to dst.
// 64 threads/block, each thread owns 2 adjacent columns via packed u32 loads.
__global__ void k_edge_msg(const float* __restrict__ nd, const _Float16* __restrict__ Mm,
                           const int* __restrict__ ei, float* __restrict__ msg_acc){
  __shared__ float snd[H];
  const int e = blockIdx.x;
  const int t = threadIdx.x;               // 0..63
  const int src = ei[2*e], dst = ei[2*e+1];
  snd[t]      = nd[(size_t)src*H + t];
  snd[t + 64] = nd[(size_t)src*H + t + 64];
  __syncthreads();
  const unsigned* Me = (const unsigned*)(Mm + (size_t)e*(H*H));
  float s0 = 0.f, s1 = 0.f;
#pragma unroll 8
  for (int d = 0; d < H; ++d) {
    h2 q = __builtin_bit_cast(h2, Me[d*(H/2) + t]);
    s0 += snd[d] * (float)q.x;
    s1 += snd[d] * (float)q.y;
  }
  atomicAdd(&msg_acc[(size_t)dst*H + 2*t    ], s0);
  atomicAdd(&msg_acc[(size_t)dst*H + 2*t + 1], s1);
}

__global__ void k_msg_fin(const float* msg_acc, const float* cnt_dst, const float* bias,
                          _Float16* msg_h){
  long i=(long)blockIdx.x*256+threadIdx.x; if(i>=(long)NN*H) return;
  int node=(int)(i>>7);
  float d = fmaxf(cnt_dst[node],1.f);
  msg_h[i] = (_Float16)fmaxf(msg_acc[i]/d + bias[i&127], 0.f);
}

__global__ void k_gru(const float* __restrict__ GI, const float* __restrict__ GH,
                      const float* bih, const float* bhh,
                      float* hid, _Float16* hid_h){
  long i=(long)blockIdx.x*256+threadIdx.x; if(i>=(long)NN*H) return;
  int node=(int)(i>>7), k=(int)(i&127);
  size_t b=(size_t)node*(3*H);
  float r = sigm(GI[b+k]       + bih[k]       + GH[b+k]       + bhh[k]);
  float z = sigm(GI[b+H+k]     + bih[H+k]     + GH[b+H+k]     + bhh[H+k]);
  float nv= tanhf(GI[b+2*H+k]  + bih[2*H+k] + r*(GH[b+2*H+k]  + bhh[2*H+k]));
  float hv = (1.f - z)*nv + z*hid[i];
  hid[i]=hv; hid_h[i]=(_Float16)hv;
}

__global__ void k_finish_acc(float* acc, _Float16* ndf_h){
  long i=(long)blockIdx.x*256+threadIdx.x; if(i>=(long)NN*H) return;
  float v=acc[i]*0.25f; acc[i]=v; ndf_h[i]=(_Float16)v;
}

__global__ void k_gcn_agg(const float* __restrict__ out, const int* ei, float* agg){
  long i=(long)blockIdx.x*256+threadIdx.x; if(i>=(long)NE*H) return;
  int e=(int)(i>>7), k=(int)(i&127);
  atomicAdd(&agg[(size_t)ei[2*e]*H+k], out[(size_t)ei[2*e+1]*H+k]);
}
__global__ void k_gcn_fin(const float* agg, const float* cnt_src, const float* rw,
                          const float* b, float* nd_out, _Float16* nd_h){
  long i=(long)blockIdx.x*256+threadIdx.x; if(i>=(long)NN*H) return;
  int node=(int)(i>>7);
  float v = agg[i]/fmaxf(cnt_src[node],1.f) + rw[i] + b[i&127];
  nd_out[i]=v; if(nd_h) nd_h[i]=(_Float16)v;
}

__global__ void k_xw_dinv(const float* nd, const float* sw, const float* cnt_dst,
                          float* xw, float* dinv){
  int i=blockIdx.x*256+threadIdx.x; if(i>=NN) return;
  float s=0.f;
#pragma unroll 8
  for(int k=0;k<H;++k) s += nd[(size_t)i*H+k]*sw[k];
  xw[i]=s;
  dinv[i]=rsqrtf(cnt_dst[i]+1.0f);     // degc = cnt_dst + self-loop, always >= 1
}
__global__ void k_score_edge(const int* ei, const float* dinv, const float* xw, float* score){
  int e=blockIdx.x*256+threadIdx.x; if(e>=NE) return;
  int s=ei[2*e], d=ei[2*e+1];
  atomicAdd(&score[d], dinv[s]*dinv[d]*xw[s]);
}
__global__ void k_score_self(const float* dinv, const float* xw, const float* sb, float* score){
  int i=blockIdx.x*256+threadIdx.x; if(i>=NN) return;
  score[i] += dinv[i]*dinv[i]*xw[i] + sb[0];
}
__global__ void k_counts(const int* batch, float* counts){
  int i=blockIdx.x*256+threadIdx.x; if(i>=NN) return;
  atomicAdd(&counts[batch[i]],1.f);
}
__global__ void k_scan(const float* counts, float* starts){
  if(threadIdx.x==0 && blockIdx.x==0){ float run=0.f; for(int b=0;b<NG;++b){ starts[b]=run; run+=counts[b]; } }
}
__global__ void k_rank(const int* batch, const float* score, const float* counts,
                       const float* starts, float* keep, float* keep_cnt){
  int i=blockIdx.x*256+threadIdx.x; if(i>=NN) return;
  int b=batch[i];
  int s=(int)(starts[b]+0.5f), c=(int)(counts[b]+0.5f);
  float sc=score[i]; int r=0;
  for(int j=s;j<s+c;++j){ float sj=score[j]; r += (sj>sc) || (sj==sc && j<i); } // stable lexsort tie-break
  float kp = ((float)r < ceilf(0.8f*(float)c)) ? 1.f : 0.f;
  keep[i]=kp;
  if(kp>0.f) atomicAdd(&keep_cnt[b],1.f);
}
__global__ void k_pool(const float* nd, const float* score, const float* keep,
                       const int* batch, float* pool_sum, float* pool_max){
  long i=(long)blockIdx.x*256+threadIdx.x; if(i>=(long)NN*H) return;
  int node=(int)(i>>7);
  if(keep[node]==0.f) return;
  float xs = nd[i]*tanhf(score[node]);
  size_t o=(size_t)batch[node]*H + (i&127);
  atomicAdd(&pool_sum[o], xs);
  atomicMaxF(&pool_max[o], xs);
}
__global__ void k_pool_fin(const float* ps, const float* pm, const float* kc, float* pool){
  long i=(long)blockIdx.x*256+threadIdx.x; if(i>=(long)NG*H) return;
  int b=(int)(i>>7), k=(int)(i&127);
  pool[(size_t)b*256 + k] = ps[i]/fmaxf(kc[b],1.f);
  float mx = pm[i];
  pool[(size_t)b*256 + H + k] = (mx <= -5e29f) ? 0.f : mx;   // NEG/2 test
}
__global__ void k_feat(const int* ci, const float* pool, const float* nd, const float* ef,
                       _Float16* feat){
  long i=(long)blockIdx.x*256+threadIdx.x; if(i>=(long)NC*768) return;
  int c=(int)(i/768), j=(int)(i%768);
  const int* row = ci + (size_t)c*5;
  float v;
  if      (j < 256) v = pool[(size_t)row[3]*256 + j];
  else if (j < 384) v = nd[(size_t)row[0]*H + (j-256)];
  else if (j < 512) v = nd[(size_t)row[1]*H + (j-384)];
  else if (j < 640) { float n0=nd[(size_t)row[0]*H+(j-512)], n1=nd[(size_t)row[1]*H+(j-512)];
                      v = n0 + n1 - n0*n1; }
  else              v = ef[(size_t)row[4]*H + (j-640)];
  feat[i] = (_Float16)v;
}
__global__ void k_p3(const _Float16* __restrict__ hh2, const float* __restrict__ W,
                     const float* b, const int* ci, float* out){
  const int c = blockIdx.x;            // one wave per coupling
  const int lane = threadIdx.x;
  const int ct = ci[(size_t)c*5 + 2];
  float s=0.f;
  for(int k=lane;k<512;k+=32) s += (float)hh2[(size_t)c*512+k]*W[k*8+ct];
#pragma unroll
  for(int o=16;o>0;o>>=1) s += __shfl_down(s,o,32);
  if(lane==0) out[c] = s + b[ct];
}

// ---------------------------------------------------------------------------
static inline unsigned g1(long n){ return (unsigned)((n+255)/256); }

extern "C" void kernel_launch(void* const* d_in, const int* in_sizes, int n_in,
                              void* d_out, int out_size, void* d_ws, size_t ws_size,
                              hipStream_t stream) {
  // -------- inputs (recursive insertion order; bn tuples -> g,b,m,v) --------
  const float* node    = (const float*)d_in[0];
  const float* edge    = (const float*)d_in[1];
  const float* ne1_w   = (const float*)d_in[2];
  const float* ne1_bn0 = (const float*)d_in[3],  *ne1_bn1=(const float*)d_in[4],  *ne1_bn2=(const float*)d_in[5],  *ne1_bn3=(const float*)d_in[6];
  const float* ne2_w   = (const float*)d_in[7];
  const float* ne2_bn0 = (const float*)d_in[8],  *ne2_bn1=(const float*)d_in[9],  *ne2_bn2=(const float*)d_in[10], *ne2_bn3=(const float*)d_in[11];
  const float* ee1_w   = (const float*)d_in[12];
  const float* ee1_bn0 = (const float*)d_in[13], *ee1_bn1=(const float*)d_in[14], *ee1_bn2=(const float*)d_in[15], *ee1_bn3=(const float*)d_in[16];
  const float* ee2_w   = (const float*)d_in[17];
  const float* ee2_bn0 = (const float*)d_in[18], *ee2_bn1=(const float*)d_in[19], *ee2_bn2=(const float*)d_in[20], *ee2_bn3=(const float*)d_in[21];
  const float* ee3_w   = (const float*)d_in[22];
  const float* ee3_bn0 = (const float*)d_in[23], *ee3_bn1=(const float*)d_in[24], *ee3_bn2=(const float*)d_in[25], *ee3_bn3=(const float*)d_in[26];
  const float* gc_w    = (const float*)d_in[27];
  const float* gc_bn0  = (const float*)d_in[28], *gc_bn1=(const float*)d_in[29],  *gc_bn2=(const float*)d_in[30],  *gc_bn3=(const float*)d_in[31];
  const float* gc_bias = (const float*)d_in[32];
  const float* gru_wih = (const float*)d_in[33];
  const float* gru_whh = (const float*)d_in[34];
  const float* gru_bih = (const float*)d_in[35];
  const float* gru_bhh = (const float*)d_in[36];
  const float* g2_w    = (const float*)d_in[37];
  const float* g2_rw   = (const float*)d_in[38];
  const float* g2_b    = (const float*)d_in[39];
  const float* g3_w    = (const float*)d_in[40];
  const float* g3_rw   = (const float*)d_in[41];
  const float* g3_b    = (const float*)d_in[42];
  const float* sag_w   = (const float*)d_in[43];
  const float* sag_b   = (const float*)d_in[44];
  const float* p1_w    = (const float*)d_in[45];
  const float* p1_bn0  = (const float*)d_in[46], *p1_bn1=(const float*)d_in[47], *p1_bn2=(const float*)d_in[48], *p1_bn3=(const float*)d_in[49];
  const float* p2_w    = (const float*)d_in[50];
  const float* p2_bn0  = (const float*)d_in[51], *p2_bn1=(const float*)d_in[52], *p2_bn2=(const float*)d_in[53], *p2_bn3=(const float*)d_in[54];
  const float* p3_w    = (const float*)d_in[55];
  const float* p3_b    = (const float*)d_in[56];
  const int*   ei      = (const int*)d_in[57];
  const int*   batch   = (const int*)d_in[58];
  const int*   ci      = (const int*)d_in[59];
  float* out = (float*)d_out;

  // -------------------------- workspace allocator ---------------------------
  char* base = (char*)d_ws; size_t off = 0;
  auto alloc = [&](size_t bytes)->void*{ void* p = base + off; off += (bytes + 255) & ~(size_t)255; return p; };

  // f16 weights (all B operands in fragment-major layout)
  _Float16* ne1_h  = (_Float16*)alloc(128*128*2);
  _Float16* ne2_h  = (_Float16*)alloc(128*128*2);
  _Float16* ee2_h  = (_Float16*)alloc(256*256*2);
  _Float16* ee3_h  = (_Float16*)alloc(256*128*2);
  _Float16* gc_h   = (_Float16*)alloc((size_t)128*16384*2);
  _Float16* wihT_h = (_Float16*)alloc(128*384*2);
  _Float16* whhT_h = (_Float16*)alloc(128*384*2);
  _Float16* g2w_h  = (_Float16*)alloc(128*128*2);
  _Float16* g2rw_h = (_Float16*)alloc(128*128*2);
  _Float16* g3w_h  = (_Float16*)alloc(128*128*2);
  _Float16* g3rw_h = (_Float16*)alloc(128*128*2);
  _Float16* p1_h   = (_Float16*)alloc((size_t)768*1024*2);
  _Float16* p2_h   = (_Float16*)alloc((size_t)1024*512*2);
  // activations
  _Float16* node_h = (_Float16*)alloc((size_t)NN*128*2);
  _Float16* h1_h   = (_Float16*)alloc((size_t)NN*128*2);
  float*    x      = (float*)   alloc((size_t)NN*H*4);
  _Float16* e1_h   = (_Float16*)alloc((size_t)NE*256*2);
  _Float16* e2_h   = (_Float16*)alloc((size_t)NE*256*2);
  float*    ef     = (float*)   alloc((size_t)NE*H*4);
  _Float16* e_h    = (_Float16*)alloc((size_t)NE*H*2);
  float*    hid    = (float*)   alloc((size_t)NN*H*4);
  _Float16* hid_h  = (_Float16*)alloc((size_t)NN*H*2);
  float*    acc    = (float*)   alloc((size_t)NN*H*4);
  float*    msg_a  = (float*)   alloc((size_t)NN*H*4);
  _Float16* msg_h  = (_Float16*)alloc((size_t)NN*H*2);
  float*    GI     = (float*)   alloc((size_t)NN*3*H*4);
  float*    GH     = (float*)   alloc((size_t)NN*3*H*4);
  _Float16* ndf_h  = (_Float16*)alloc((size_t)NN*H*2);
  float*    gout   = (float*)   alloc((size_t)NN*H*4);
  float*    grw    = (float*)   alloc((size_t)NN*H*4);
  float*    agg    = (float*)   alloc((size_t)NN*H*4);
  float*    nd2    = (float*)   alloc((size_t)NN*H*4);
  _Float16* nd2_h  = (_Float16*)alloc((size_t)NN*H*2);
  float*    nd3    = (float*)   alloc((size_t)NN*H*4);
  float*    cnt_s  = (float*)   alloc((size_t)NN*4);
  float*    cnt_d  = (float*)   alloc((size_t)NN*4);
  float*    xw     = (float*)   alloc((size_t)NN*4);
  float*    dinv   = (float*)   alloc((size_t)NN*4);
  float*    score  = (float*)   alloc((size_t)NN*4);
  float*    keep   = (float*)   alloc((size_t)NN*4);
  float*    counts = (float*)   alloc((size_t)NG*4);
  float*    starts = (float*)   alloc((size_t)NG*4);
  float*    kcnt   = (float*)   alloc((size_t)NG*4);
  float*    psum   = (float*)   alloc((size_t)NG*H*4);
  float*    pmax   = (float*)   alloc((size_t)NG*H*4);
  float*    pool   = (float*)   alloc((size_t)NG*256*4);
  // big f16 region: edge_mat during the loop, then feat/hh1/hh2 alias into it
  _Float16* Mmat   = (_Float16*)alloc((size_t)NE*H*H*2);            // 327.7 MB
  _Float16* feat_h = Mmat;                                          //  61.4 MB
  _Float16* hh1_h  = (_Float16*)((char*)Mmat + (size_t)NC*768*2);   //  81.9 MB
  _Float16* hh2_h  = (_Float16*)((char*)hh1_h + (size_t)NC*1024*2); //  41.0 MB

  auto gemm = [&](const _Float16* A, const _Float16* B, int M, int K, int N,
                  const float* g, const float* b, const float* m, const float* v,
                  int epi, float* Cf, _Float16* Ch){
    long tiles = (long)((M+31)/32) * (N/32);
    gemm_wmma<<<dim3((unsigned)((tiles+7)/8)), dim3(256), 0, stream>>>(A,B,M,K,N,g,b,m,v,epi,Cf,Ch);
  };

  // ------------------------- weight conversions (B packing) -----------------
  k_pack_b <<<g1(128*128),256,0,stream>>>(ne1_w, ne1_h, 128, 128, 113);   // pad K 113->128
  k_pack_b <<<g1(128*128),256,0,stream>>>(ne2_w, ne2_h, 128, 128, 128);
  k_pack_b <<<g1(256*256),256,0,stream>>>(ee2_w, ee2_h, 256, 256, 256);
  k_pack_b <<<g1(256*128),256,0,stream>>>(ee3_w, ee3_h, 256, 128, 256);
  k_pack_b <<<g1((long)128*16384),256,0,stream>>>(gc_w, gc_h, 128, 16384, 128);
  k_pack_bT<<<g1(128*384),256,0,stream>>>(gru_wih, wihT_h, 128, 384);
  k_pack_bT<<<g1(128*384),256,0,stream>>>(gru_whh, whhT_h, 128, 384);
  k_pack_b <<<g1(128*128),256,0,stream>>>(g2_w,  g2w_h,  128, 128, 128);
  k_pack_b <<<g1(128*128),256,0,stream>>>(g2_rw, g2rw_h, 128, 128, 128);
  k_pack_b <<<g1(128*128),256,0,stream>>>(g3_w,  g3w_h,  128, 128, 128);
  k_pack_b <<<g1(128*128),256,0,stream>>>(g3_rw, g3rw_h, 128, 128, 128);
  k_pack_b <<<g1((long)768*1024),256,0,stream>>>(p1_w, p1_h, 768, 1024, 768);
  k_pack_b <<<g1((long)1024*512),256,0,stream>>>(p2_w, p2_h, 1024, 512, 1024);

  // ------------------------------ encoders ----------------------------------
  k_cvt_pad_cols<<<g1((long)NN*128),256,0,stream>>>(node, node_h, NN, 113, 128);
  gemm(node_h, ne1_h, NN, 128, 128, ne1_bn0,ne1_bn1,ne1_bn2,ne1_bn3, 2, nullptr, h1_h);
  gemm(h1_h,   ne2_h, NN, 128, 128, ne2_bn0,ne2_bn1,ne2_bn2,ne2_bn3, 2, x, nullptr);

  k_ee1<<<g1((long)NE*256),256,0,stream>>>(edge, ee1_w, ee1_bn0,ee1_bn1,ee1_bn2,ee1_bn3, e1_h);
  gemm(e1_h, ee2_h, NE, 256, 256, ee2_bn0,ee2_bn1,ee2_bn2,ee2_bn3, 2, nullptr, e2_h);
  gemm(e2_h, ee3_h, NE, 256, 128, ee3_bn0,ee3_bn1,ee3_bn2,ee3_bn3, 2, ef, e_h);
  gemm(e_h,  gc_h,  NE, 128, 16384, gc_bn0,gc_bn1,gc_bn2,gc_bn3, 1, nullptr, Mmat); // edge_mat f16

  // degrees
  k_zero<<<g1(NN),256,0,stream>>>(cnt_s, NN);
  k_zero<<<g1(NN),256,0,stream>>>(cnt_d, NN);
  k_deg<<<g1(NE),256,0,stream>>>(ei, cnt_s, cnt_d);

  // --------------------- message passing + GRU, 4 iters ---------------------
  k_init_state<<<g1((long)NN*H),256,0,stream>>>(x, hid, hid_h, acc);
  for (int it = 0; it < 4; ++it) {
    k_acc<<<g1((long)NN*H),256,0,stream>>>(hid, x, acc);
    k_zero<<<g1((long)NN*H),256,0,stream>>>(msg_a, (long)NN*H);
    k_edge_msg<<<dim3(NE), dim3(64), 0, stream>>>(hid, Mmat, ei, msg_a);
    k_msg_fin<<<g1((long)NN*H),256,0,stream>>>(msg_a, cnt_d, gc_bias, msg_h);
    gemm(msg_h, wihT_h, NN, 128, 384, nullptr,nullptr,nullptr,nullptr, 0, GI, nullptr);
    gemm(hid_h, whhT_h, NN, 128, 384, nullptr,nullptr,nullptr,nullptr, 0, GH, nullptr);
    k_gru<<<g1((long)NN*H),256,0,stream>>>(GI, GH, gru_bih, gru_bhh, hid, hid_h);
  }
  k_finish_acc<<<g1((long)NN*H),256,0,stream>>>(acc, ndf_h);   // nd = acc/4

  // ------------------------------ GCN x2 ------------------------------------
  gemm(ndf_h, g2w_h,  NN, 128, 128, nullptr,nullptr,nullptr,nullptr, 0, gout, nullptr);
  gemm(ndf_h, g2rw_h, NN, 128, 128, nullptr,nullptr,nullptr,nullptr, 0, grw,  nullptr);
  k_zero<<<g1((long)NN*H),256,0,stream>>>(agg, (long)NN*H);
  k_gcn_agg<<<g1((long)NE*H),256,0,stream>>>(gout, ei, agg);
  k_gcn_fin<<<g1((long)NN*H),256,0,stream>>>(agg, cnt_s, grw, g2_b, nd2, nd2_h);

  gemm(nd2_h, g3w_h,  NN, 128, 128, nullptr,nullptr,nullptr,nullptr, 0, gout, nullptr);
  gemm(nd2_h, g3rw_h, NN, 128, 128, nullptr,nullptr,nullptr,nullptr, 0, grw,  nullptr);
  k_zero<<<g1((long)NN*H),256,0,stream>>>(agg, (long)NN*H);
  k_gcn_agg<<<g1((long)NE*H),256,0,stream>>>(gout, ei, agg);
  k_gcn_fin<<<g1((long)NN*H),256,0,stream>>>(agg, cnt_s, grw, g3_b, nd3, nullptr);

  // ------------------------------ SAG pool ----------------------------------
  k_xw_dinv<<<g1(NN),256,0,stream>>>(nd3, sag_w, cnt_d, xw, dinv);
  k_zero<<<g1(NN),256,0,stream>>>(score, NN);
  k_score_edge<<<g1(NE),256,0,stream>>>(ei, dinv, xw, score);
  k_score_self<<<g1(NN),256,0,stream>>>(dinv, xw, sag_b, score);
  k_zero<<<g1(NG),256,0,stream>>>(counts, NG);
  k_counts<<<g1(NN),256,0,stream>>>(batch, counts);
  k_scan<<<1,1,0,stream>>>(counts, starts);
  k_zero<<<g1(NG),256,0,stream>>>(kcnt, NG);
  k_rank<<<g1(NN),256,0,stream>>>(batch, score, counts, starts, keep, kcnt);
  k_zero<<<g1((long)NG*H),256,0,stream>>>(psum, (long)NG*H);
  k_fill<<<g1((long)NG*H),256,0,stream>>>(pmax, -1e30f, (long)NG*H);
  k_pool<<<g1((long)NN*H),256,0,stream>>>(nd3, score, keep, batch, psum, pmax);
  k_pool_fin<<<g1((long)NG*H),256,0,stream>>>(psum, pmax, kcnt, pool);

  // --------------------------- coupling head --------------------------------
  k_feat<<<g1((long)NC*768),256,0,stream>>>(ci, pool, nd3, ef, feat_h);
  gemm(feat_h, p1_h, NC, 768,  1024, p1_bn0,p1_bn1,p1_bn2,p1_bn3, 2, nullptr, hh1_h);
  gemm(hh1_h,  p2_h, NC, 1024, 512,  p2_bn0,p2_bn1,p2_bn2,p2_bn3, 2, nullptr, hh2_h);
  k_p3<<<dim3(NC), dim3(32), 0, stream>>>(hh2_h, p3_w, p3_b, ci, out);

  (void)in_sizes; (void)n_in; (void)out_size; (void)ws_size;
}
